// AE_st_80650895884832
// MI455X (gfx1250) — compile-verified, matrix operand order checked by reading, not run on hardware
//
#include <hip/hip_runtime.h>
#include <hip/hip_bf16.h>
#include <math.h>

// ---------------------------------------------------------------------------
// Problem constants (match reference)
// ---------------------------------------------------------------------------
#define NN    10000
#define GENE  3000
#define EMB   32
#define HID_E 512

// GEMM tiling
#define BM 128
#define BN 128
#define BK 32

typedef _Float16 v16h  __attribute__((ext_vector_type(16)));
typedef _Float16 h8    __attribute__((ext_vector_type(8)));
typedef _Float16 h4    __attribute__((ext_vector_type(4)));
typedef float    v8f   __attribute__((ext_vector_type(8)));
typedef float    f32x4 __attribute__((ext_vector_type(4)));

union Frag16 { v16h v; h8 h[2]; };

enum { ACT_NONE = 0, ACT_RELU = 1, ACT_SIGMOID = 2, ACT_SOFTPLUS = 3, ACT_EXP = 4 };

__device__ __forceinline__ float act_apply(float v, int act) {
    switch (act) {
        case ACT_RELU:     return v > 0.f ? v : 0.f;
        case ACT_SIGMOID:  return 1.f / (1.f + __expf(-v));
        case ACT_SOFTPLUS: return v > 20.f ? v : log1pf(__expf(v));
        case ACT_EXP:      return __expf(v);
        default:           return v;
    }
}

// Native CDNA5 float atomic max (global_atomic_max_num_f32, no-return form).
__device__ __forceinline__ void atomicMaxF32(float* addr, float val) {
    asm volatile("global_atomic_max_num_f32 %0, %1, off"
                 :: "v"((unsigned long long)(uintptr_t)addr), "v"(val)
                 : "memory");
}

// ---------------------------------------------------------------------------
// Tiled WMMA GEMM: C[M,N] = act(A[M,K] @ B[K,N] + bias[N])
// A,B,C fp32 in global; fp32->fp16 conversion fused into staging.
// 256 threads = 8 waves; wave (wm,wn) owns a 32x64 C tile = 2x4 WMMA frags.
// Interior blocks: unguarded global_load_b128 staging + double-buffered LDS.
// Boundary blocks / K-tail: guarded scalar path.
// ---------------------------------------------------------------------------
__global__ __launch_bounds__(256) void gemm_wmma_f16(
    const float* __restrict__ A, const float* __restrict__ B,
    const float* __restrict__ bias, float* __restrict__ C,
    int M, int N, int K, int act)
{
    __shared__ _Float16 sA[2][BM * BK];   // [m][k] row-major
    __shared__ _Float16 sB[2][BN * BK];   // [n][k]  (B staged transposed)

    const int t    = threadIdx.x;
    const int bm   = blockIdx.x * BM;
    const int bn   = blockIdx.y * BN;
    const int wid  = t >> 5;
    const int lane = t & 31;
    const int wm   = (wid & 3) * 32;     // wave row offset inside tile
    const int wn   = (wid >> 2) * 64;    // wave col offset inside tile
    const int lr   = lane & 15;
    const int sel  = lane >> 4;          // selects K-half per ISA A-layout

    v8f acc[2][4] = {};

    const bool interior = (bm + BM <= M) && (bn + BN <= N);
    const int  Kmain    = interior ? (K & ~(BK - 1)) : 0;
    const int  ntile    = Kmain / BK;

    f32x4 areg[4], breg[4];

    auto load_tile = [&](int k0) {
        #pragma unroll
        for (int i = 0; i < 4; ++i) {                 // A: 128x32 = 1024 f32x4
            int id = t + i * 256;
            int r = id >> 3, c4 = id & 7;
            areg[i] = *(const f32x4*)(A + (size_t)(bm + r) * K + k0 + c4 * 4);
        }
        #pragma unroll
        for (int i = 0; i < 4; ++i) {                 // B: 32x128, coalesced over n
            int id = t + i * 256;
            int kk = id >> 5, n4 = id & 31;
            breg[i] = *(const f32x4*)(B + (size_t)(k0 + kk) * N + bn + n4 * 4);
        }
    };

    auto store_lds = [&](int buf) {
        #pragma unroll
        for (int i = 0; i < 4; ++i) {
            int id = t + i * 256;
            int r = id >> 3, c4 = id & 7;
            h4 hv;
            #pragma unroll
            for (int j = 0; j < 4; ++j) hv[j] = (_Float16)areg[i][j];
            *(h4*)(&sA[buf][r * BK + c4 * 4]) = hv;   // ds_store_b64
        }
        #pragma unroll
        for (int i = 0; i < 4; ++i) {
            int id = t + i * 256;
            int kk = id >> 5, n4 = id & 31;
            #pragma unroll
            for (int j = 0; j < 4; ++j)               // transpose into [n][k]
                sB[buf][(n4 * 4 + j) * BK + kk] = (_Float16)breg[i][j];
        }
    };

    auto compute = [&](int buf) {
        // Fragment gathers (two 16B LDS reads each) per ISA 16-bit A layout:
        // lane<16 -> K {0..7,16..23}, lane>=16 -> K {8..15,24..31}
        Frag16 af[2], bf[4];
        #pragma unroll
        for (int mi = 0; mi < 2; ++mi) {
            int row = wm + mi * 16 + lr;
            const h8* p = (const h8*)(&sA[buf][row * BK + sel * 8]);
            af[mi].h[0] = p[0];
            af[mi].h[1] = p[2];
        }
        #pragma unroll
        for (int ni = 0; ni < 4; ++ni) {
            int col = wn + ni * 16 + lr;
            const h8* p = (const h8*)(&sB[buf][col * BK + sel * 8]);
            bf[ni].h[0] = p[0];
            bf[ni].h[1] = p[2];
        }
        #pragma unroll
        for (int mi = 0; mi < 2; ++mi)
            #pragma unroll
            for (int ni = 0; ni < 4; ++ni)
                acc[mi][ni] = __builtin_amdgcn_wmma_f32_16x16x32_f16(
                    false, af[mi].v, false, bf[ni].v,
                    (short)0, acc[mi][ni], false, false);
    };

    // ---- main double-buffered loop (interior blocks) ----
    int cur = 0;
    if (ntile > 0) {
        load_tile(0);
        store_lds(cur);
        __syncthreads();
        for (int kt = 0; kt < ntile; ++kt) {
            const bool more = (kt + 1 < ntile);
            if (more) load_tile((kt + 1) * BK);   // global loads in flight ...
            compute(cur);                         // ... behind the WMMAs
            if (more) {
                store_lds(cur ^ 1);
                __syncthreads();
                cur ^= 1;
            }
        }
    }

    // ---- guarded path: boundary blocks and K tail ----
    for (int k0 = Kmain; k0 < K; k0 += BK) {
        __syncthreads();
        for (int i = 0; i < 16; ++i) {
            int idx = t + i * 256;
            int r = idx >> 5, c = idx & 31;
            int gr = bm + r, gc = k0 + c;
            float v = (gr < M && gc < K) ? A[(size_t)gr * K + gc] : 0.f;
            sA[0][idx] = (_Float16)v;
        }
        for (int i = 0; i < 16; ++i) {
            int idx = t + i * 256;
            int kk = idx >> 7, nn = idx & 127;
            int gk = k0 + kk, gn = bn + nn;
            float v = (gk < K && gn < N) ? B[(size_t)gk * N + gn] : 0.f;
            sB[0][nn * BK + kk] = (_Float16)v;
        }
        __syncthreads();
        compute(0);
    }

    // ---- epilogue: C/D layout -> row = r + sel*8, col = lane%16 ----
    if (interior) {
        #pragma unroll
        for (int mi = 0; mi < 2; ++mi) {
            #pragma unroll
            for (int ni = 0; ni < 4; ++ni) {
                int col = bn + wn + ni * 16 + lr;
                float bv = (bias != nullptr) ? bias[col] : 0.f;
                #pragma unroll
                for (int r = 0; r < 8; ++r) {
                    int row = bm + wm + mi * 16 + sel * 8 + r;
                    C[(size_t)row * N + col] = act_apply(acc[mi][ni][r] + bv, act);
                }
            }
        }
    } else {
        #pragma unroll
        for (int mi = 0; mi < 2; ++mi) {
            #pragma unroll
            for (int ni = 0; ni < 4; ++ni) {
                int col = bn + wn + ni * 16 + lr;
                float bv = (bias != nullptr && col < N) ? bias[col] : 0.f;
                #pragma unroll
                for (int r = 0; r < 8; ++r) {
                    int row = bm + wm + mi * 16 + sel * 8 + r;
                    if (row < M && col < N)
                        C[(size_t)row * N + col] = act_apply(acc[mi][ni][r] + bv, act);
                }
            }
        }
    }
}

// ---------------------------------------------------------------------------
// a_s[row] = dot(H[row,:], a_src),  a_d[row] = dot(H[row,:], a_dst)
// One wave32 per row, shuffle reduction.
// ---------------------------------------------------------------------------
__global__ __launch_bounds__(256) void rowdot2_k(
    const float* __restrict__ H, const float* __restrict__ asrc,
    const float* __restrict__ adst, float* __restrict__ outs,
    float* __restrict__ outd, int n, int F)
{
    int wid = threadIdx.x >> 5, lane = threadIdx.x & 31;
    int row = blockIdx.x * 8 + wid;
    if (row >= n) return;
    const float* hr = H + (size_t)row * F;
    float s = 0.f, d = 0.f;
    for (int f = lane; f < F; f += 32) {
        float h = hr[f];
        s += h * asrc[f];
        d += h * adst[f];
    }
    #pragma unroll
    for (int o = 16; o > 0; o >>= 1) {
        s += __shfl_xor(s, o);
        d += __shfl_xor(d, o);
    }
    if (lane == 0) { outs[row] = s; outd[row] = d; }
}

__global__ void fill_k(float* __restrict__ p, float v, int n) {
    int i = blockIdx.x * blockDim.x + threadIdx.x;
    if (i < n) p[i] = v;
}

// out[i,f] = bias[f] (pre-seeds "+ b" of the GAT output); F is a power of two
__global__ void initout_k(float* __restrict__ out, const float* __restrict__ b,
                          int total, int Fmask) {
    int i = blockIdx.x * blockDim.x + threadIdx.x;
    if (i < total) out[i] = b[i & Fmask];
}

__global__ void elu_k(float* __restrict__ p, int n) {
    int i = blockIdx.x * blockDim.x + threadIdx.x;
    if (i < n) {
        float v = p[i];
        p[i] = v > 0.f ? v : __expf(v) - 1.f;
    }
}

__global__ void combine_k(float* __restrict__ e, const float* __restrict__ a,
                          const float* __restrict__ b, int n) {
    int i = blockIdx.x * blockDim.x + threadIdx.x;
    if (i < n) e[i] = 0.5f * a[i] + 0.5f * b[i];
}

// Pass 1: segment max of leaky_relu(a_s[src]+a_d[dst]) over dst
__global__ void edge_max_k(const int* __restrict__ src, const int* __restrict__ dst,
                           const float* __restrict__ as, const float* __restrict__ ad,
                           float* __restrict__ m, int E, int n) {
    int t = blockIdx.x * blockDim.x + threadIdx.x;
    int total = E + n;                         // self loops appended
    if (t >= total) return;
    int s = t < E ? src[t] : (t - E);
    int d = t < E ? dst[t] : (t - E);
    float e = as[s] + ad[d];
    e = e > 0.f ? e : 0.2f * e;
    atomicMaxF32(m + d, e);
}

// Pass 2: ex = exp(e - m[dst]); denom[dst] += ex
__global__ void edge_ex_k(const int* __restrict__ src, const int* __restrict__ dst,
                          const float* __restrict__ as, const float* __restrict__ ad,
                          const float* __restrict__ m, float* __restrict__ denom,
                          float* __restrict__ exb, int E, int n) {
    int t = blockIdx.x * blockDim.x + threadIdx.x;
    int total = E + n;
    if (t >= total) return;
    int s = t < E ? src[t] : (t - E);
    int d = t < E ? dst[t] : (t - E);
    float e = as[s] + ad[d];
    e = e > 0.f ? e : 0.2f * e;
    float ex = __expf(e - m[d]);
    exb[t] = ex;
    atomicAdd(denom + d, ex);
}

// Pass 3: out[dst,:] += (ex/denom[dst]) * H[src,:]   (one block per edge)
__global__ void edge_agg_k(const int* __restrict__ src, const int* __restrict__ dst,
                           const float* __restrict__ exb, const float* __restrict__ denom,
                           const float* __restrict__ H, float* __restrict__ out,
                           int E, int n, int F) {
    int e = blockIdx.x;
    int total = E + n;
    if (e >= total) return;
    int s = e < E ? src[e] : (e - E);
    int d = e < E ? dst[e] : (e - E);
    float alpha = exb[e] / denom[d];
    const float* hs = H + (size_t)s * F;
    float* od = out + (size_t)d * F;
    for (int f = threadIdx.x; f < F; f += blockDim.x)
        atomicAdd(od + f, alpha * hs[f]);
}

// ---------------------------------------------------------------------------
// Host orchestration
// ---------------------------------------------------------------------------
static inline int ceil_div(int a, int b) { return (a + b - 1) / b; }

extern "C" void kernel_launch(void* const* d_in, const int* in_sizes, int n_in,
                              void* d_out, int out_size, void* d_ws, size_t ws_size,
                              hipStream_t stream) {
    (void)n_in; (void)out_size; (void)ws_size;

    const float* x     = (const float*)d_in[0];
    const int*   e_exp = (const int*)d_in[1];
    const int*   e_sp  = (const int*)d_in[2];
    const int E_exp = in_sizes[1] / 2;
    const int E_sp  = in_sizes[2] / 2;

    // JAX pytree flatten order: nested dict keys sorted -> W, a_dst, a_src, b
    // ge block: [W1, a_dst1, a_src1, b1, W2, a_dst2, a_src2, b2]
    // dec block: [W0, b0, W1, b1, W2, b2]
    const int GE1 = 3, GE2 = 11, DPI = 19, DMU = 25, DTH = 31;

    // Workspace carve-up (fp32)
    float* ws  = (float*)d_ws;
    float* H1  = ws;                         // N x 512  (x @ W1)
    float* O1  = H1 + (size_t)NN * HID_E;    // N x 512  (aggregated, then ELU)
    float* H2  = O1 + (size_t)NN * HID_E;    // N x 32   (O1 @ W2)
    float* as_ = H2 + (size_t)NN * EMB;      // N
    float* ad_ = as_ + NN;                   // N
    float* m_  = ad_ + NN;                   // N
    float* dn_ = m_  + NN;                   // N
    float* exb = dn_ + NN;                   // max edges+loops = 210000
    // decoders reuse H1 (N x 256) and O1 (N x 512)
    float* D1 = H1;
    float* D2 = O1;

    float* out      = (float*)d_out;
    float* emb      = out;
    float* emb_expr = out + (size_t)NN * EMB;
    float* emb_sp   = out + 2 * (size_t)NN * EMB;
    float* pi_o     = out + 3 * (size_t)NN * EMB;
    float* mu_o     = pi_o + (size_t)NN * GENE;
    float* th_o     = mu_o + (size_t)NN * GENE;

    auto gemm = [&](const float* A, const float* B, const float* bias, float* C,
                    int M, int Nc, int K, int act) {
        dim3 g(ceil_div(M, BM), ceil_div(Nc, BN));
        gemm_wmma_f16<<<g, dim3(256), 0, stream>>>(A, B, bias, C, M, Nc, K, act);
    };

    auto gat_attn = [&](const float* H, const float* asrc, const float* adst,
                        const float* bvec, const int* esrc, const int* edst,
                        int E, float* O, int F) {
        int total = E + NN;
        rowdot2_k<<<ceil_div(NN, 8), 256, 0, stream>>>(H, asrc, adst, as_, ad_, NN, F);
        fill_k<<<ceil_div(NN, 256), 256, 0, stream>>>(m_, -__builtin_inff(), NN);
        fill_k<<<ceil_div(NN, 256), 256, 0, stream>>>(dn_, 0.f, NN);
        initout_k<<<ceil_div(NN * F, 256), 256, 0, stream>>>(O, bvec, NN * F, F - 1);
        edge_max_k<<<ceil_div(total, 256), 256, 0, stream>>>(esrc, edst, as_, ad_, m_, E, NN);
        edge_ex_k<<<ceil_div(total, 256), 256, 0, stream>>>(esrc, edst, as_, ad_, m_, dn_, exb, E, NN);
        int bs = (F >= 256) ? 256 : ((F + 31) & ~31);
        edge_agg_k<<<total, bs, 0, stream>>>(esrc, edst, exb, dn_, H, O, E, NN, F);
    };

    auto encoder = [&](int gb, const int* ei, int E, float* emb_out) {
        const float* W1 = (const float*)d_in[gb + 0];
        const float* ad1 = (const float*)d_in[gb + 1];
        const float* as1 = (const float*)d_in[gb + 2];
        const float* b1 = (const float*)d_in[gb + 3];
        const float* W2 = (const float*)d_in[gb + 4];
        const float* ad2 = (const float*)d_in[gb + 5];
        const float* as2 = (const float*)d_in[gb + 6];
        const float* b2 = (const float*)d_in[gb + 7];
        const int* esrc = ei;
        const int* edst = ei + E;

        gemm(x, W1, nullptr, H1, NN, HID_E, GENE, ACT_NONE);         // h = x @ W1
        gat_attn(H1, as1, ad1, b1, esrc, edst, E, O1, HID_E);        // layer 1 attn
        elu_k<<<ceil_div(NN * HID_E, 256), 256, 0, stream>>>(O1, NN * HID_E);
        gemm(O1, W2, nullptr, H2, NN, EMB, HID_E, ACT_NONE);         // h = elu @ W2
        gat_attn(H2, as2, ad2, b2, esrc, edst, E, emb_out, EMB);     // layer 2 attn
    };

    auto decoder = [&](int db, int final_act, float* out_ptr) {
        const float* W0 = (const float*)d_in[db + 0];
        const float* b0 = (const float*)d_in[db + 1];
        const float* W1 = (const float*)d_in[db + 2];
        const float* b1 = (const float*)d_in[db + 3];
        const float* W2 = (const float*)d_in[db + 4];
        const float* b2 = (const float*)d_in[db + 5];
        gemm(emb, W0, b0, D1, NN, 256, EMB, ACT_RELU);
        gemm(D1, W1, b1, D2, NN, 512, 256, ACT_RELU);
        gemm(D2, W2, b2, out_ptr, NN, GENE, 512, final_act);
    };

    // spatial encoder uses ge1 + spatial edges; expression uses ge2 + expr edges
    encoder(GE1, e_sp, E_sp, emb_sp);
    encoder(GE2, e_exp, E_exp, emb_expr);
    combine_k<<<ceil_div(NN * EMB, 256), 256, 0, stream>>>(emb, emb_sp, emb_expr, NN * EMB);

    decoder(DPI, ACT_SIGMOID, pi_o);
    decoder(DMU, ACT_SOFTPLUS, mu_o);
    decoder(DTH, ACT_EXP, th_o);
}
